// HeteroGCN_32762010534126
// MI455X (gfx1250) — matrix operand order repaired and checked by reading, not compiled, not run on hardware
//
#include <hip/hip_runtime.h>
#include <hip/hip_bf16.h>

// ---------------------------------------------------------------------------
// HeteroGCN (3-layer hetero-SAGE, 11 node types, 17 relations, N=100000,
// D=128, E=1.6M per relation) for MI455X / gfx1250 (wave32, WMMA bf16).
// All workspace tensors padded to Npad=100032 rows so the WMMA GEMM has
// zero predicated memory ops.
// ---------------------------------------------------------------------------

typedef __attribute__((ext_vector_type(16))) __bf16 v16bf;
typedef __attribute__((ext_vector_type(8)))  float  v8f;

#define GNN_N       100000
#define GNN_NPAD    100032   // 1563 * 64
#define GNN_D       128
#define GNN_E       1600000
#define GNN_TYPES   11
#define GNN_REL     17
#define GNN_LAYERS  3

// 1 / rel_counts (bincount of DST, all >= 1)
__constant__ float K_RELINV[GNN_TYPES] =
    {0.2f, 1.f, 0.5f, 1.f, 0.5f, 1.f, 1.f, 1.f, 1.f, 1.f, 1.f};

// -------------------------------- utils ------------------------------------

__device__ __forceinline__ float wave_sum32(float v) {
#pragma unroll
  for (int m = 16; m > 0; m >>= 1) v += __shfl_xor(v, m, 32);
  return v;
}

__device__ __forceinline__ void atom_add_f32(float* p, float v) {
  __hip_atomic_fetch_add(p, v, __ATOMIC_RELAXED, __HIP_MEMORY_SCOPE_AGENT);
}

// ---------------- 0) weight prep: f32 [k][n] -> bf16 [n][k] -----------------
__global__ void hg_wt_prep(const float* __restrict__ w,
                           __bf16* __restrict__ wt,
                           int total) {
  const int o = blockIdx.x * blockDim.x + threadIdx.x;
  if (o >= total) return;
  const int mat = o >> 14;            // / 16384
  const int rem = o & 16383;
  const int nn = rem >> 7;            // output row (original column)
  const int kk = rem & 127;           // output col (original row)
  wt[o] = (__bf16)w[((size_t)mat << 14) + ((size_t)kk << 7) + nn];
}

// ------------------------- 1) edge gather + scatter -------------------------
// One wave32 per edge; lane L handles features [4L, 4L+4).
__global__ void hg_scatter(const float* __restrict__ xsrc,
                           const long long* __restrict__ src_idx,
                           const long long* __restrict__ dst_idx,
                           float* __restrict__ agg,
                           float* __restrict__ deg,
                           int nEdges) {
  const int lane = threadIdx.x & 31;
  const int wv   = threadIdx.x >> 5;
  const long long e = (long long)blockIdx.x * 8 + wv;
  if (e >= nEdges) return;
  const long long s = src_idx[e];
  const long long d = dst_idx[e];
  const float4 v = *(const float4*)(xsrc + s * GNN_D + lane * 4);
  float* base = agg + d * GNN_D + lane * 4;
  atom_add_f32(base + 0, v.x);
  atom_add_f32(base + 1, v.y);
  atom_add_f32(base + 2, v.z);
  atom_add_f32(base + 3, v.w);
  if (lane == 0) atom_add_f32(deg + d, 1.0f);
}

// ------------------- 2) fused SAGE GEMM (WMMA bf16 -> f32) ------------------
// sums += (agg/max(deg,1)) @ Wl + xroot @ Wr + bias
// Block = 256 threads = 8 waves; block owns 64 rows x 128 cols.
// Wave w owns 16-col stripe; loops 4 row tiles of 16.
// Wlt/Wrt are bf16, pre-transposed to [n][k] so B fragments are contiguous.
// agg/deg/sums are padded to >= blockIdx range; deg/agg/xroot loads all use the clamped row m.
__global__ void hg_sage_gemm(const float* __restrict__ agg,
                             const float* __restrict__ deg,
                             const float* __restrict__ xroot,
                             const __bf16* __restrict__ Wlt,
                             const __bf16* __restrict__ Wrt,
                             const float* __restrict__ bias,
                             float* __restrict__ sums,
                             int nRows) {
  const int lane = threadIdx.x & 31;
  const int wv   = threadIdx.x >> 5;
  const int nloc = lane & 15;          // column within stripe's 16
  const int hhi  = lane >> 4;          // 0 = low half lanes, 1 = high half
  const int n    = (wv << 4) + nloc;   // output column 0..127

  // B fragments (32x16 bf16): lanes 0-15 hold K=kb..kb+15, lanes 16-31 hold
  // K=kb+16..kb+31 -> with [n][k] layout this is one contiguous 32B load.
  v16bf bl[4], br[4];
#pragma unroll
  for (int kc = 0; kc < 4; ++kc) {
    const size_t off = ((size_t)n << 7) + (kc << 5) + (hhi << 4);
    bl[kc] = *(const v16bf*)(Wlt + off);
    br[kc] = *(const v16bf*)(Wrt + off);
  }
  const float bn = bias[n];

#pragma unroll
  for (int rt = 0; rt < 4; ++rt) {
    const int R = blockIdx.x * 64 + rt * 16;
    const int m = min(R + nloc, nRows - 1);   // clamp: always-valid loads
    const float invd = 1.0f / fmaxf(deg[m], 1.0f);
    const float* aggRow = agg   + ((size_t)m << 7);
    const float* xr     = xroot + ((size_t)m << 7);

    v8f c = {bn, bn, bn, bn, bn, bn, bn, bn};

#pragma unroll
    for (int kc = 0; kc < 4; ++kc) {
      // A fragment, this code's contiguous k order: elements 0..7 -> K=k0..k0+7,
      // elements 8..15 -> K=k0+8..k0+15, k0 = kc*32 + 16*hhi.
      const int k0 = (kc << 5) + (hhi << 4);
      const v8f qa_lo = *(const v8f*)(aggRow + k0);
      const v8f qa_hi = *(const v8f*)(aggRow + k0 + 8);
      const v8f qx_lo = *(const v8f*)(xr + k0);
      const v8f qx_hi = *(const v8f*)(xr + k0 + 8);
      v16bf a1, a2;
#pragma unroll
      for (int j = 0; j < 8; ++j) {
        a1[j]     = (__bf16)(qa_lo[j] * invd);
        a1[8 + j] = (__bf16)(qa_hi[j] * invd);
        a2[j]     = (__bf16)qx_lo[j];
        a2[8 + j] = (__bf16)qx_hi[j];
      }
      c = __builtin_amdgcn_wmma_f32_16x16x32_bf16(
              false, a1, false, bl[kc], (short)0, c, false, false);
      c = __builtin_amdgcn_wmma_f32_16x16x32_bf16(
              false, a2, false, br[kc], (short)0, c, false, false);
    }

    // D layout (f32 16x16): lane holds column n; VGPR v -> row R + 8*hhi + v.
    // sums is padded: stores always in-bounds, no guards.
    float* p = sums + (((size_t)(R + (hhi << 3))) << 7) + n;
#pragma unroll
    for (int v = 0; v < 8; ++v) {
      p[(size_t)v << 7] += c[v];
    }
  }
}

// --------- 3) per-row: /rel_count, leaky_relu(0.01), LayerNorm (inplace) ----
// grid = (N/8, 11); one wave32 per row; lane handles 4 features.
__global__ void hg_finalize(float* __restrict__ x,
                            const float* __restrict__ gamma,
                            const float* __restrict__ beta) {
  const int lane = threadIdx.x & 31;
  const int t = blockIdx.y;
  const int i = blockIdx.x * 8 + (threadIdx.x >> 5);
  const float inv = K_RELINV[t];

  float* rp = x + (((size_t)t * GNN_NPAD + i) << 7) + lane * 4;
  float4 v = *(const float4*)rp;
  v.x *= inv; v.y *= inv; v.z *= inv; v.w *= inv;
  v.x = v.x >= 0.f ? v.x : 0.01f * v.x;
  v.y = v.y >= 0.f ? v.y : 0.01f * v.y;
  v.z = v.z >= 0.f ? v.z : 0.01f * v.z;
  v.w = v.w >= 0.f ? v.w : 0.01f * v.w;

  const float mean = wave_sum32(v.x + v.y + v.z + v.w) * (1.0f / GNN_D);
  const float dx = v.x - mean, dy = v.y - mean, dz = v.z - mean, dw = v.w - mean;
  const float var =
      wave_sum32(dx * dx + dy * dy + dz * dz + dw * dw) * (1.0f / GNN_D);
  const float rs = rsqrtf(var + 1e-5f);

  const float4 gm = *(const float4*)(gamma + lane * 4);
  const float4 bt = *(const float4*)(beta + lane * 4);
  float4 o;
  o.x = dx * rs * gm.x + bt.x;
  o.y = dy * rs * gm.y + bt.y;
  o.z = dz * rs * gm.z + bt.z;
  o.w = dw * rs * gm.w + bt.w;
  *(float4*)rp = o;
}

// ---------------- 4) head: cat(11*128) @ fc_w + fc_b, sigmoid ---------------
// One wave32 per node.
__global__ void hg_head(const float* __restrict__ x,
                        const float* __restrict__ fcw,
                        const float* __restrict__ fcb,
                        float* __restrict__ out) {
  const int lane = threadIdx.x & 31;
  const int nid = blockIdx.x * 8 + (threadIdx.x >> 5);
  float acc = 0.f;
#pragma unroll
  for (int t = 0; t < GNN_TYPES; ++t) {
    const float4 xv =
        *(const float4*)(x + (((size_t)t * GNN_NPAD + nid) << 7) + lane * 4);
    const float4 wv = *(const float4*)(fcw + t * GNN_D + lane * 4);
    acc += xv.x * wv.x + xv.y * wv.y + xv.z * wv.z + xv.w * wv.w;
  }
  acc = wave_sum32(acc);
  if (lane == 0) {
    const float z = acc + fcb[0];
    out[nid] = 1.0f / (1.0f + __expf(-z));
  }
}

// --------------------------------- launch -----------------------------------

static const int SRC_H[GNN_REL] = {0, 1, 0, 2, 1, 0, 3, 3, 0, 4, 0, 1, 0, 0, 5, 0, 0};
static const int DST_H[GNN_REL] = {1, 0, 2, 0, 2, 3, 0, 8, 4, 0, 7, 4, 6, 5, 0, 9, 10};

extern "C" void kernel_launch(void* const* d_in, const int* in_sizes, int n_in,
                              void* d_out, int out_size, void* d_ws, size_t ws_size,
                              hipStream_t stream) {
  (void)in_sizes; (void)n_in; (void)out_size; (void)ws_size;

  const float*     x_all = (const float*)d_in[0];
  const long long* ei    = (const long long*)d_in[1];
  const float*     w_l   = (const float*)d_in[2];
  const float*     w_r   = (const float*)d_in[3];
  const float*     bvec  = (const float*)d_in[4];
  const float*     lng   = (const float*)d_in[5];
  const float*     lnb   = (const float*)d_in[6];
  const float*     fcw   = (const float*)d_in[7];
  const float*     fcb   = (const float*)d_in[8];
  float*           out   = (float*)d_out;

  const size_t slab   = (size_t)GNN_NPAD * GNN_D;            // padded slab elems
  const size_t xpadB  = (size_t)GNN_TYPES * slab * 4;        // padded x bytes
  const size_t wtElems = (size_t)GNN_LAYERS * GNN_REL * GNN_D * GNN_D; // 835584

  char* ws = (char*)d_ws;
  float*  buf0 = (float*)ws;
  float*  buf1 = (float*)(ws + xpadB);
  float*  agg  = (float*)(ws + 2 * xpadB);
  float*  deg  = (float*)(ws + 2 * xpadB + slab * 4);
  __bf16* wtl  = (__bf16*)(ws + 2 * xpadB + slab * 4 + (size_t)GNN_NPAD * 4);
  __bf16* wtr  = wtl + wtElems;

  const dim3 blk(256);
  // 0) pre-transpose + bf16-convert all 102 weight matrices (3.3 MB total)
  hg_wt_prep<<<dim3((wtElems + 255) / 256), blk, 0, stream>>>(w_l, wtl, (int)wtElems);
  hg_wt_prep<<<dim3((wtElems + 255) / 256), blk, 0, stream>>>(w_r, wtr, (int)wtElems);

  const dim3 gridScatter(GNN_E / 8);          // 8 edges (waves) / block
  const dim3 gridGemm(GNN_NPAD / 64);         // 1563 blocks, 64 rows each
  const dim3 gridFin(GNN_N / 8, GNN_TYPES);   // one wave per real row
  const dim3 gridHead(GNN_N / 8);

  const float* xin = x_all;
  bool xinPadded = false;
  for (int L = 0; L < GNN_LAYERS; ++L) {
    float* sums = (L & 1) ? buf1 : buf0;
    hipMemsetAsync(sums, 0, xpadB, stream);
    for (int r = 0; r < GNN_REL; ++r) {
      hipMemsetAsync(agg, 0, slab * 4, stream);
      hipMemsetAsync(deg, 0, (size_t)GNN_NPAD * 4, stream);
      const int s = SRC_H[r], d = DST_H[r];
      const size_t sOff = (size_t)s * (xinPadded ? slab : (size_t)GNN_N * GNN_D);
      const size_t dOff = (size_t)d * (xinPadded ? slab : (size_t)GNN_N * GNN_D);
      hg_scatter<<<gridScatter, blk, 0, stream>>>(
          xin + sOff,
          ei + (size_t)(2 * r) * GNN_E,
          ei + (size_t)(2 * r + 1) * GNN_E,
          agg, deg, GNN_E);
      const size_t wi = ((size_t)L * GNN_REL + r);
      hg_sage_gemm<<<gridGemm, blk, 0, stream>>>(
          agg, deg,
          xin + dOff,
          wtl + (wi << 14),
          wtr + (wi << 14),
          bvec + wi * GNN_D,
          sums + (size_t)d * slab,
          GNN_N);
    }
    hg_finalize<<<gridFin, blk, 0, stream>>>(
        sums, lng + (size_t)L * GNN_D, lnb + (size_t)L * GNN_D);
    xin = sums;        // LayerNorm'd in place; padded layout from here on
    xinPadded = true;
  }

  // After L=0,1,2 the final activations live in buf0 (padded layout).
  hg_head<<<gridHead, blk, 0, stream>>>(buf0, fcw, fcb, out);
}
